// MissingGAT_43095701848697
// MI455X (gfx1250) — compile-verified, matrix operand order checked by reading, not run on hardware
//
#include <hip/hip_runtime.h>

typedef _Float16 h16;
typedef __attribute__((ext_vector_type(16))) _Float16 v16h;
typedef __attribute__((ext_vector_type(8)))  _Float16 v8h;
typedef __attribute__((ext_vector_type(8)))  float    v8f;
typedef __attribute__((ext_vector_type(4)))  float    v4f;

__device__ __forceinline__ v8f wmma_f16(v16h a, v16h b, v8f c) {
  // D = A(16x32 f16) * B(32x16 f16) + C(16x16 f32)
  return __builtin_amdgcn_wmma_f32_16x16x32_f16(false, a, false, b, (short)0, c, false, false);
}
__device__ __forceinline__ v16h cat16(v8h a, v8h b) {
  return __builtin_shufflevector(a, b, 0,1,2,3,4,5,6,7,8,9,10,11,12,13,14,15);
}

// ---------------- adjacency bitmask ----------------
__global__ __launch_bounds__(256) void adj_zero_kernel(unsigned* __restrict__ adj) {
  adj[blockIdx.x * 256 + threadIdx.x] = 0u;   // 2048 words total
}

__global__ __launch_bounds__(256) void adj_set_kernel(const int* __restrict__ ei, int E,
                                                      unsigned* __restrict__ adj) {
  int e = blockIdx.x * 256 + threadIdx.x;
  if (e < E) {
    int q = ei[e];        // edge_index[0][e]  (mask row)
    int k = ei[E + e];    // edge_index[1][e]  (mask col)
    atomicOr(&adj[q * 8 + (k >> 5)], 1u << (k & 31));
  }
}

// ---------------- fused QKV projection ----------------
// x[65536][128] f32 -> Q,K fp16 [row][32] (Q pre-scaled by 1/sqrt(D)),
// V transposed fp16 Vt[b][h][d][n]
__global__ __launch_bounds__(256) void qkv_kernel(
    const float* __restrict__ x,
    const float* __restrict__ Wq, const float* __restrict__ bq,
    const float* __restrict__ Wk, const float* __restrict__ bk,
    const float* __restrict__ Wv, const float* __restrict__ bv,
    h16* __restrict__ Qb, h16* __restrict__ Kb, h16* __restrict__ Vt)
{
  // Weights pre-swizzled into per-lane WMMA B-fragment order:
  // sWB[mat][kk][nt][lane][frag], frag = chunk*8 + j  (two b128 reads per fragment)
  __shared__ __align__(16) h16 sWB[3][4][2][32][16];
  __shared__ float sB[3][32];
  for (int i = threadIdx.x; i < 12288; i += 256) {
    const int mat = i >> 12;            // 4096 elements per matrix
    const int r   = i & 4095;           // r = k*32 + n
    const int k = r >> 5, n = r & 31;
    const int kk = k >> 5, kr = k & 31;
    const int nt = n >> 4, mm = n & 15;
    const int chunk = kr >> 4;          // 0: K k0+{0..15}, 1: K k0+{16..31}
    const int hb = (kr >> 3) & 1;       // lane half
    const int j = kr & 7;
    const float* W = (mat == 0) ? Wq : (mat == 1) ? Wk : Wv;
    sWB[mat][kk][nt][hb * 16 + mm][chunk * 8 + j] = (h16)W[r];
  }
  if (threadIdx.x < 32) {
    sB[0][threadIdx.x] = bq[threadIdx.x];
    sB[1][threadIdx.x] = bk[threadIdx.x];
    sB[2][threadIdx.x] = bv[threadIdx.x];
  }
  __syncthreads();

  const int lane = threadIdx.x & 31, wave = threadIdx.x >> 5;
  const int tile = blockIdx.x * 8 + wave;        // 4096 tiles of 16 rows
  const int row0 = tile * 16;
  const int m = lane & 15, hi = lane >> 4;

  const v4f* xv = (const v4f*)(x + (size_t)(row0 + m) * 128);

  v8f acc[3][2];
  #pragma unroll
  for (int a = 0; a < 3; ++a)
    #pragma unroll
    for (int b = 0; b < 2; ++b) { v8f z = {}; acc[a][b] = z; }

  #pragma unroll
  for (int kk = 0; kk < 4; ++kk) {
    const int f0 = kk * 8 + hi * 2;     // float4 index of k0 + hi*8
    const v4f xa = xv[f0], xb = xv[f0 + 1];
    const v4f xc = xv[f0 + 4], xd = xv[f0 + 5];   // k0 + 16 + hi*8
    v16h A;
    #pragma unroll
    for (int j = 0; j < 4; ++j) {
      A[j]      = (h16)xa[j];
      A[4 + j]  = (h16)xb[j];
      A[8 + j]  = (h16)xc[j];
      A[12 + j] = (h16)xd[j];
    }
    #pragma unroll
    for (int mat = 0; mat < 3; ++mat) {
      #pragma unroll
      for (int nt = 0; nt < 2; ++nt) {
        const v8h* bp = (const v8h*)&sWB[mat][kk][nt][lane][0];
        acc[mat][nt] = wmma_f16(A, cat16(bp[0], bp[1]), acc[mat][nt]);
      }
    }
  }

  const float rsD = 0.35355339059327373f;  // 1/sqrt(8) folded into Q
  #pragma unroll
  for (int nt = 0; nt < 2; ++nt) {
    const int c = nt * 16 + m;
    #pragma unroll
    for (int r = 0; r < 8; ++r) {
      const int grow = row0 + r + 8 * hi;
      const float qv = acc[0][nt][r] + sB[0][c];
      const float kv = acc[1][nt][r] + sB[1][c];
      const float vv = acc[2][nt][r] + sB[2][c];
      Qb[(size_t)grow * 32 + c] = (h16)(qv * rsD);
      Kb[(size_t)grow * 32 + c] = (h16)kv;
      const int b = grow >> 8, nn = grow & 255;
      Vt[(size_t)((b * 4 + (c >> 3)) * 8 + (c & 7)) * 256 + nn] = (h16)vv;
    }
  }
}

// ---------------- masked attention (one wave per (b,h,16 q-rows)) ----------------
__global__ __launch_bounds__(64) void attn_kernel(
    const h16* __restrict__ Qb, const h16* __restrict__ Kb, const h16* __restrict__ Vt,
    const unsigned* __restrict__ adj, h16* __restrict__ AO)
{
  __shared__ __align__(16) float    sS[2][16][256];   // score strip per wave
  __shared__ __align__(16) h16      sP[2][16][256];   // softmax probs (A-operand source)
  __shared__ unsigned sA[2][16][8];                   // adjacency bits for 16 query rows
  __shared__ float    sInv[2][16];                    // 1/rowsum

  const int lane = threadIdx.x & 31, wave = threadIdx.x >> 5;
  const int task = blockIdx.x * 2 + wave;             // 16384 = B*H*16
  const int qt = task & 15;
  const int bh = task >> 4;
  const int b = bh >> 2, h = bh & 3;
  const int q0 = qt * 16;
  const int m = lane & 15, hi = lane >> 4;
  const v8h z8 = {};

  // stage adjacency rows (128 words)
  #pragma unroll
  for (int i = 0; i < 4; ++i) {
    const int idx = lane * 4 + i;
    sA[wave][idx >> 3][idx & 7] = adj[(q0 + (idx >> 3)) * 8 + (idx & 7)];
  }

  // Q fragment: K-dim is d (8 real, zero-padded to 32) -> only hi==0 chunk0 live
  const v8h q8 = *(const v8h*)(Qb + (size_t)(b * 256 + q0 + m) * 32 + h * 8);
  const v16h Aq = cat16(hi == 0 ? q8 : z8, z8);

  asm volatile("s_wait_dscnt 0" ::: "memory");

  // scores: 16 k-tiles of 16 columns
  for (int t = 0; t < 16; ++t) {
    const v8h k8 = *(const v8h*)(Kb + (size_t)(b * 256 + t * 16 + m) * 32 + h * 8);
    const v16h Bk = cat16(hi == 0 ? k8 : z8, z8);
    v8f c = {};
    c = wmma_f16(Aq, Bk, c);
    const int col = t * 16 + m;
    const unsigned bit = 1u << (col & 31);
    const int widx = col >> 5;
    #pragma unroll
    for (int r = 0; r < 8; ++r) {
      const int row = r + 8 * hi;
      sS[wave][row][col] = (sA[wave][row][widx] & bit) ? c[r] : -__builtin_inff();
    }
  }

  asm volatile("s_wait_dscnt 0" ::: "memory");

  // row softmax: lane -> (row = m, half = hi covers 128 columns), vectorized x4
  {
    const int row = m;
    const v4f* sv = (const v4f*)&sS[wave][row][hi * 128];
    v4f m4 = sv[0];
    for (int j = 1; j < 32; ++j) m4 = __builtin_elementwise_max(m4, sv[j]);
    float mx = fmaxf(fmaxf(m4[0], m4[1]), fmaxf(m4[2], m4[3]));
    mx = fmaxf(mx, __shfl_xor(mx, 16, 32));      // combine halves
    float sum = 0.f;
    for (int j = 0; j < 16; ++j) {
      const v4f a = sv[2 * j], c = sv[2 * j + 1];
      v8h e;
      #pragma unroll
      for (int i = 0; i < 4; ++i) {
        const float ea = __expf(a[i] - mx);      // exp(-inf - mx) = 0
        const float eb = __expf(c[i] - mx);
        sum += ea + eb;
        e[i] = (h16)ea;
        e[4 + i] = (h16)eb;
      }
      *(v8h*)&sP[wave][row][hi * 128 + j * 8] = e;
    }
    sum += __shfl_xor(sum, 16, 32);
    if (hi == 0) sInv[wave][row] = 1.0f / sum;   // self-loops guarantee sum > 0
  }

  asm volatile("s_wait_dscnt 0" ::: "memory");

  // attn x V: 8 WMMAs over K=256 k-nodes; B columns = d (8 real, padded)
  v8f acc = {};
  const h16* vtb = Vt + (size_t)((b * 4 + h) * 8) * 256;
  const h16* vrow = vtb + (size_t)(m & 7) * 256;     // clamped; zero-selected below
  #pragma unroll
  for (int cch = 0; cch < 8; ++cch) {
    const int kb = cch * 32;
    const v8h* pr = (const v8h*)&sP[wave][m][kb];
    const v16h Ap = cat16(pr[hi], pr[2 + hi]);
    const v8h* vv = (const v8h*)(vrow + kb);
    const v8h b0 = vv[hi], b1 = vv[2 + hi];
    const v16h Bv = cat16(m < 8 ? b0 : z8, m < 8 ? b1 : z8);
    acc = wmma_f16(Ap, Bv, acc);
  }

  if (m < 8) {
    #pragma unroll
    for (int r = 0; r < 8; ++r) {
      const int row = r + 8 * hi;
      const float v = acc[r] * sInv[wave][row];
      AO[((size_t)(b * 256 + q0 + row) * 4 + h) * 8 + m] = (h16)v;
    }
  }
}

// ---------------- output projection: AO[65536][32] @ Wo[32][8] + bo ----------------
__global__ __launch_bounds__(256) void outproj_kernel(
    const h16* __restrict__ AO, const float* __restrict__ Wo, const float* __restrict__ bo,
    float* __restrict__ out)
{
  __shared__ __align__(16) h16 sWoB[32][16];   // Wo in per-lane B-fragment order
  __shared__ float sBo[8];
  for (int i = threadIdx.x; i < 512; i += 256) ((h16*)sWoB)[i] = (h16)0.f;
  __syncthreads();
  if (threadIdx.x < 256) {
    const int k = threadIdx.x >> 3, n = threadIdx.x & 7;   // Wo[k][n]
    const int chunk = k >> 4, hb = (k >> 3) & 1, j = k & 7;
    sWoB[hb * 16 + n][chunk * 8 + j] = (h16)Wo[k * 8 + n];
  }
  if (threadIdx.x < 8) sBo[threadIdx.x] = bo[threadIdx.x];
  __syncthreads();

  const int lane = threadIdx.x & 31, wave = threadIdx.x >> 5;
  const int tile = blockIdx.x * 8 + wave;      // 4096 tiles
  const int row0 = tile * 16;
  const int m = lane & 15, hi = lane >> 4;

  const v8h* av = (const v8h*)(AO + (size_t)(row0 + m) * 32);
  const v16h A = cat16(av[hi], av[2 + hi]);

  const v8h* wv = (const v8h*)&sWoB[lane][0];
  const v16h Bf = cat16(wv[0], wv[1]);

  v8f C = {};
  const float bb = (m < 8) ? sBo[m] : 0.f;
  #pragma unroll
  for (int r = 0; r < 8; ++r) C[r] = bb;       // bias pre-loaded into C operand

  const v8f D = wmma_f16(A, Bf, C);
  if (m < 8) {
    #pragma unroll
    for (int r = 0; r < 8; ++r) out[(size_t)(row0 + r + 8 * hi) * 8 + m] = D[r];
  }
}

extern "C" void kernel_launch(void* const* d_in, const int* in_sizes, int n_in,
                              void* d_out, int out_size, void* d_ws, size_t ws_size,
                              hipStream_t stream)
{
  const float* x  = (const float*)d_in[0];
  const int*   ei = (const int*)d_in[1];
  const float* Wq = (const float*)d_in[2];
  const float* bq = (const float*)d_in[3];
  const float* Wk = (const float*)d_in[4];
  const float* bk = (const float*)d_in[5];
  const float* Wv = (const float*)d_in[6];
  const float* bv = (const float*)d_in[7];
  const float* Wo = (const float*)d_in[8];
  const float* bo = (const float*)d_in[9];
  float* out = (float*)d_out;
  const int E = in_sizes[1] / 2;   // edge_index is [2, E]

  char* ws = (char*)d_ws;
  const size_t MB = 1024 * 1024;
  unsigned* adj = (unsigned*)ws;                        // 8 KB
  h16* Qb = (h16*)(ws + 8192);                          // 4 MB  (65536*32 fp16)
  h16* Kb = (h16*)(ws + 8192 + 4 * MB);                 // 4 MB
  h16* Vt = (h16*)(ws + 8192 + 8 * MB);                 // 4 MB  ([b][h][d][n])
  h16* AO = (h16*)(ws + 8192 + 12 * MB);                // 4 MB  ([b][n][h][d])

  adj_zero_kernel<<<8, 256, 0, stream>>>(adj);
  adj_set_kernel<<<(E + 255) / 256, 256, 0, stream>>>(ei, E, adj);
  qkv_kernel<<<512, 256, 0, stream>>>(x, Wq, bq, Wk, bk, Wv, bv, Qb, Kb, Vt);
  attn_kernel<<<8192, 64, 0, stream>>>(Qb, Kb, Vt, adj, AO);
  outproj_kernel<<<512, 256, 0, stream>>>(AO, Wo, bo, out);
}
